// EfficientMoEFFN_5188320494403
// MI455X (gfx1250) — matrix operand order repaired and checked by reading, not compiled, not run on hardware
//
#include <hip/hip_runtime.h>
#include <stdint.h>

typedef __attribute__((ext_vector_type(16))) __bf16 v16bf;
typedef __attribute__((ext_vector_type(8)))  float  v8f;
typedef unsigned short u16t;
typedef unsigned int   u32t;

#define C_DIM 1024
#define H_DIM 4096
#define E_DIM 8
#define N_TOK 4096
#define HC    128      // H chunk width
#define TM    32       // tokens per tile

__device__ __forceinline__ u16t f2bf(float f) {
  u32t x = __float_as_uint(f);
  u32t r = x + 0x7fffu + ((x >> 16) & 1u);   // round-to-nearest-even
  return (u16t)(r >> 16);
}

union V16U { v16bf bf; uint4 q[2]; };

// A fragment (16x32 bf16, ISA 7.12.2): lane holds row m=L%16, two contiguous
// 8-element K chunks at k0+half*8 and k0+16+half*8.
__device__ __forceinline__ v16bf load_fragA(const u16t* p) {
  V16U t;
  t.q[0] = *(const uint4*)(p);
  t.q[1] = *(const uint4*)(p + 16);
  return t.bf;
}
// B fragment: pre-packed contiguous 16 elements per lane.
__device__ __forceinline__ v16bf load_fragB(const u16t* p) {
  V16U t;
  t.q[0] = *(const uint4*)(p);
  t.q[1] = *(const uint4*)(p + 8);
  return t.bf;
}
__device__ __forceinline__ v8f wmma_bf16(v16bf a, v16bf b, v8f c) {
  return __builtin_amdgcn_wmma_f32_16x16x32_bf16(false, a, false, b, (short)0, c, false, false);
}

__global__ void moe_zero_cnt(int* cnt) {
  if (threadIdx.x < E_DIM) cnt[threadIdx.x] = 0;
}

// Pack fp32 weights [E, Kdim, Ndim] -> bf16 WMMA B-fragments.
// dst layout: e*perExpert + (nTile*kChunks + kChunk)*512 + L*16 + i
// element (L,i): n = nTile*16 + (L&15), k = kChunk*32 + (i>>3)*16 + (L>>4)*8 + (i&7)
__global__ __launch_bounds__(256) void moe_pack_w(
    const float* __restrict__ src, u16t* __restrict__ dst,
    int kChunks, int Ndim, int Kdim, int perExpert)
{
  int idx = blockIdx.x * 256 + threadIdx.x;
  int e   = idx / perExpert;
  int rem = idx - e * perExpert;
  int frag = rem >> 9;
  int le   = rem & 511;
  int L = le >> 4, i = le & 15;
  int nTile  = frag / kChunks;
  int kChunk = frag - nTile * kChunks;
  int n = nTile * 16 + (L & 15);
  int k = kChunk * 32 + (i >> 3) * 16 + (L >> 4) * 8 + (i & 7);
  dst[idx] = f2bf(src[((size_t)e * Kdim + k) * Ndim + n]);
}

// One wave32 per token: gate = x@Wg + bg, top-2, softmax, routing lists.
// Also converts the token's x row to bf16.
__global__ __launch_bounds__(256) void moe_gate(
    const float* __restrict__ x, const float* __restrict__ Wg,
    const float* __restrict__ bg, u16t* __restrict__ xbf,
    int* __restrict__ cnt, int* __restrict__ tokDst, float* __restrict__ tokProb)
{
  int wave = threadIdx.x >> 5;
  int lane = threadIdx.x & 31;
  int t = blockIdx.x * 8 + wave;
  if (t >= N_TOK) return;

  float acc[E_DIM];
  #pragma unroll
  for (int e = 0; e < E_DIM; ++e) acc[e] = 0.f;

  const float* xr = x + (size_t)t * C_DIM;
  for (int c = lane; c < C_DIM; c += 32) {
    float xv = xr[c];
    xbf[(size_t)t * C_DIM + c] = f2bf(xv);
    const float* wr = Wg + (size_t)c * E_DIM;
    #pragma unroll
    for (int e = 0; e < E_DIM; ++e) acc[e] += xv * wr[e];
  }
  #pragma unroll
  for (int off = 16; off >= 1; off >>= 1) {
    #pragma unroll
    for (int e = 0; e < E_DIM; ++e) acc[e] += __shfl_down(acc[e], off, 32);
  }
  if (lane == 0) {
    float g[E_DIM];
    #pragma unroll
    for (int e = 0; e < E_DIM; ++e) g[e] = acc[e] + bg[e];
    int i0 = 0; float v0 = g[0];
    #pragma unroll
    for (int e = 1; e < E_DIM; ++e) if (g[e] > v0) { v0 = g[e]; i0 = e; }
    int i1 = -1; float v1 = -3.4e38f;
    #pragma unroll
    for (int e = 0; e < E_DIM; ++e) if (e != i0 && g[e] > v1) { v1 = g[e]; i1 = e; }
    float p0 = 1.f / (1.f + __expf(v1 - v0));   // softmax over {v0, v1}
    float p1 = 1.f - p0;
    int s0 = atomicAdd(&cnt[i0], 1);
    tokDst[i0 * N_TOK + s0] = 2 * t;     tokProb[i0 * N_TOK + s0] = p0;
    int s1 = atomicAdd(&cnt[i1], 1);
    tokDst[i1 * N_TOK + s1] = 2 * t + 1; tokProb[i1 * N_TOK + s1] = p1;
  }
}

// Fused expert FFN for one (expert, 32-token tile).
// 8 waves; wave w owns 16 H-cols of each chunk (GEMM1) and C-cols
// [w*128, w*128+128) of the output (GEMM2, 16 v8f accumulators).
__global__ __launch_bounds__(256, 1) void moe_ffn(
    const u16t* __restrict__ xbf, const u16t* __restrict__ w1f,
    const u16t* __restrict__ w2f, const float* __restrict__ b1,
    const float* __restrict__ b2, const int* __restrict__ cnt,
    const int* __restrict__ tokDst, const float* __restrict__ tokProb,
    float* __restrict__ yc)
{
  __shared__ u16t x_lds[TM * C_DIM];   // 64 KB bf16 token tile
  __shared__ u16t h_lds[TM * HC];      //  8 KB bf16 hidden chunk
  __shared__ float probs_s[TM];
  __shared__ int   dst_s[TM];

  int e = blockIdx.y;
  int nTok = cnt[e];
  int base = blockIdx.x * TM;
  if (base >= nTok) return;
  int rows = nTok - base; if (rows > TM) rows = TM;

  int tid  = threadIdx.x;
  int w    = tid >> 5;
  int lane = tid & 31;
  int n16  = lane & 15;
  int half = lane >> 4;

  if (tid < TM) {
    int d = 0; float p = 0.f;
    if (tid < rows) {
      d = tokDst[e * N_TOK + base + tid];
      p = tokProb[e * N_TOK + base + tid];
    }
    dst_s[tid] = d; probs_s[tid] = p;
  }
  __syncthreads();

  { // gather 32 token rows of bf16 x into LDS (8 threads per row, 16B loads)
    int r = tid >> 3, sub = tid & 7;
    const uint4* srcp = (const uint4*)(xbf + (size_t)(dst_s[r] >> 1) * C_DIM);
    uint4* dstp = (uint4*)(x_lds + r * C_DIM);
    #pragma unroll
    for (int i = 0; i < 16; ++i) dstp[sub * 16 + i] = srcp[sub * 16 + i];
  }
  __syncthreads();

  v8f accY[16];
  #pragma unroll
  for (int i = 0; i < 16; ++i) accY[i] = (v8f){0.f,0.f,0.f,0.f,0.f,0.f,0.f,0.f};

  for (int c = 0; c < H_DIM / HC; ++c) {
    // ---- GEMM1: h[32, HC] = relu(x @ W1[:, chunk] + b1) ----
    v8f aH0 = (v8f){0.f,0.f,0.f,0.f,0.f,0.f,0.f,0.f};
    v8f aH1 = (v8f){0.f,0.f,0.f,0.f,0.f,0.f,0.f,0.f};
    const u16t* w1p    = w1f + (((size_t)e * (H_DIM/16) + c * 8 + w) * (C_DIM/32)) * 512 + lane * 16;
    const u16t* a0base = x_lds + n16 * C_DIM + half * 8;
    const u16t* a1base = a0base + 16 * C_DIM;
    #pragma unroll 4
    for (int kc = 0; kc < C_DIM / 32; ++kc) {
      v16bf B  = load_fragB(w1p + kc * 512);
      v16bf A0 = load_fragA(a0base + kc * 32);
      v16bf A1 = load_fragA(a1base + kc * 32);
      aH0 = wmma_bf16(A0, B, aH0);
      aH1 = wmma_bf16(A1, B, aH1);
    }
    float b1v = b1[e * H_DIM + c * HC + w * 16 + n16];
    #pragma unroll
    for (int v = 0; v < 8; ++v) {
      int m0 = v + 8 * half;
      float h0 = aH0[v] + b1v; h0 = h0 > 0.f ? h0 : 0.f;
      h_lds[m0 * HC + w * 16 + n16] = f2bf(h0);
      float h1 = aH1[v] + b1v; h1 = h1 > 0.f ? h1 : 0.f;
      h_lds[(m0 + 16) * HC + w * 16 + n16] = f2bf(h1);
    }
    __syncthreads();

    // ---- GEMM2: y[32, C] += h_chunk @ W2[chunk, :] ----
    #pragma unroll
    for (int kk = 0; kk < HC / 32; ++kk) {
      const u16t* hp = h_lds + n16 * HC + kk * 32 + half * 8;
      v16bf A0 = load_fragA(hp);
      v16bf A1 = load_fragA(hp + 16 * HC);
      #pragma unroll
      for (int nt = 0; nt < 8; ++nt) {
        const u16t* w2p = w2f + (((size_t)e * (C_DIM/16) + w * 8 + nt) * (H_DIM/32) + (c * 4 + kk)) * 512 + lane * 16;
        v16bf B = load_fragB(w2p);
        accY[nt]     = wmma_bf16(A0, B, accY[nt]);
        accY[8 + nt] = wmma_bf16(A1, B, accY[8 + nt]);
      }
    }
    __syncthreads();
  }

  // epilogue: bias, scale by routed prob, write unique (token,k) slot
  #pragma unroll
  for (int nt = 0; nt < 8; ++nt) {
    int col = w * 128 + nt * 16 + n16;
    float b2v = b2[e * C_DIM + col];
    #pragma unroll
    for (int ms = 0; ms < 2; ++ms) {
      #pragma unroll
      for (int v = 0; v < 8; ++v) {
        int m = ms * 16 + v + 8 * half;
        if (m < rows) {
          float val = (accY[ms * 8 + nt][v] + b2v) * probs_s[m];
          yc[(size_t)dst_s[m] * C_DIM + col] = val;
        }
      }
    }
  }
}

__global__ __launch_bounds__(256) void moe_combine(
    const float* __restrict__ yc, float* __restrict__ out)
{
  int idx = blockIdx.x * 256 + threadIdx.x;   // < N_TOK*C_DIM
  int t = idx >> 10, cc = idx & (C_DIM - 1);
  out[idx] = yc[(size_t)(2 * t) * C_DIM + cc] + yc[(size_t)(2 * t + 1) * C_DIM + cc];
}

extern "C" void kernel_launch(void* const* d_in, const int* in_sizes, int n_in,
                              void* d_out, int out_size, void* d_ws, size_t ws_size,
                              hipStream_t stream) {
  const float* x  = (const float*)d_in[0];
  const float* Wg = (const float*)d_in[1];
  const float* bg = (const float*)d_in[2];
  const float* W1 = (const float*)d_in[3];
  const float* b1 = (const float*)d_in[4];
  const float* W2 = (const float*)d_in[5];
  const float* b2 = (const float*)d_in[6];
  float* out = (float*)d_out;
  (void)in_sizes; (void)n_in; (void)out_size; (void)ws_size;

  char* ws = (char*)d_ws;
  size_t off = 0;
  auto carve = [&](size_t bytes) {
    char* p = ws + off; off += (bytes + 255) & ~(size_t)255; return p;
  };
  u16t*  w1f     = (u16t*)carve((size_t)E_DIM * C_DIM * H_DIM * 2);   // 64 MB
  u16t*  w2f     = (u16t*)carve((size_t)E_DIM * H_DIM * C_DIM * 2);   // 64 MB
  u16t*  xbf     = (u16t*)carve((size_t)N_TOK * C_DIM * 2);           //  8 MB
  float* yc      = (float*)carve((size_t)N_TOK * 2 * C_DIM * 4);      // 32 MB
  int*   cnt     = (int*)carve(256);
  int*   tokDst  = (int*)carve((size_t)E_DIM * N_TOK * 4);
  float* tokProb = (float*)carve((size_t)E_DIM * N_TOK * 4);

  const int perExpert = (H_DIM / 16) * (C_DIM / 32) * 512;  // 4,194,304 (same for W2)

  moe_zero_cnt<<<1, 32, 0, stream>>>(cnt);
  moe_pack_w<<<(E_DIM * perExpert) / 256, 256, 0, stream>>>(W1, w1f, C_DIM / 32, H_DIM, C_DIM, perExpert);
  moe_pack_w<<<(E_DIM * perExpert) / 256, 256, 0, stream>>>(W2, w2f, H_DIM / 32, C_DIM, H_DIM, perExpert);
  moe_gate<<<N_TOK / 8, 256, 0, stream>>>(x, Wg, bg, xbf, cnt, tokDst, tokProb);
  moe_ffn<<<dim3(N_TOK / TM, E_DIM), 256, 0, stream>>>(xbf, w1f, w2f, b1, b2, cnt, tokDst, tokProb, yc);
  moe_combine<<<(N_TOK * C_DIM) / 256, 256, 0, stream>>>(yc, out);
}
